// FractalDecoderV6_19645180412086
// MI455X (gfx1250) — matrix operand order repaired and checked by reading, not compiled, not run on hardware
//
#include <hip/hip_runtime.h>
#include <stdint.h>

typedef __attribute__((ext_vector_type(16))) __bf16 v16bf;
typedef __attribute__((ext_vector_type(8)))  float  v8f;

#define BATCH  16384
#define LVLS   8
#define EDIM   8
#define HSIZE  16384
#define KFREQ  32
#define HIDD   256
#define FFD    512
#define TSTEPS 48
#define LATD   32
#define NVOCAB 256
#define INDIM  161
#define INPAD  192
#define MB     64      // rows per block in the fractal loop kernel

// ---------- small helpers ----------
__device__ __forceinline__ uint16_t f2bf(float f) {
    uint32_t u = __float_as_uint(f);
    u += 0x7FFFu + ((u >> 16) & 1u);      // round-to-nearest-even
    return (uint16_t)(u >> 16);
}
__device__ __forceinline__ float gelu_exact(float x) {
    return 0.5f * x * (1.0f + erff(x * 0.70710678118654752f));
}

// A fragment: 16x32 bf16, rows row0..row0+15, cols kk..kk+31 of row-major [*, strideK].
// Contiguous k order used here (same as B): lanes 0-15 -> M=lane, VGPR v: K = (lane/16)*16 + v*2 (pairs).
__device__ __forceinline__ v16bf load_a_frag(const uint16_t* p, int row0, int strideK, int kk) {
    int lane = threadIdx.x & 31;
    int lh   = lane >> 4;
    int m    = lane & 15;
    union { v16bf v; uint32_t u[8]; } r;
    const uint16_t* rowp = p + (size_t)(row0 + m) * strideK + kk;
#pragma unroll
    for (int v = 0; v < 8; ++v) {
        int k = (lh << 4) + (v << 1);
        r.u[v] = *(const uint32_t*)(rowp + k);
    }
    return r.v;
}

// B fragment: 32x16 bf16 from W^T stored [N][K] row-major; cols n0..n0+15, K rows kk..kk+31.
// Contiguous k order used here (same as A): lane = N (mod 16), lanes 0-15 hold K=kk..kk+15, lanes 16-31 hold K=kk+16..kk+31.
__device__ __forceinline__ v16bf load_b_frag(const uint16_t* p, int n0, int strideK, int kk) {
    int lane = threadIdx.x & 31;
    int lh   = lane >> 4;
    int n    = lane & 15;
    union { v16bf v; uint32_t u[8]; } r;
    const uint16_t* colp = p + (size_t)(n0 + n) * strideK + kk + (lh << 4);
#pragma unroll
    for (int v = 0; v < 8; ++v) r.u[v] = *(const uint32_t*)(colp + 2 * v);
    return r.v;
}

// ---------- prep: fp32 [K][N] -> bf16 [N][Kpad] (transposed, zero-padded K) ----------
__global__ void wtrans_kernel(const float* __restrict__ W, uint16_t* __restrict__ Wt,
                              int K, int N, int Kpad) {
    int idx = blockIdx.x * blockDim.x + threadIdx.x;
    if (idx >= N * Kpad) return;
    int n = idx / Kpad, k = idx - n * Kpad;
    float v = (k < K) ? W[(size_t)k * N + n] : 0.0f;
    Wt[(size_t)n * Kpad + k] = f2bf(v);
}

// ---------- features: [B, INPAD] bf16 ----------
__global__ void features_kernel(const float* __restrict__ x, const float* __restrict__ z,
                                const float* __restrict__ ht, const int* __restrict__ res,
                                const float* __restrict__ freqs, uint16_t* __restrict__ feats) {
    int b = blockIdx.x * blockDim.x + threadIdx.x;
    if (b >= BATCH) return;
    float xn = fminf(fmaxf(x[b], 0.0f), 1.0f);
    uint16_t* f = feats + (size_t)b * INPAD;
    f[0] = f2bf(xn);
    const float twopi = 6.28318530717958647f;
#pragma unroll
    for (int k = 0; k < KFREQ; ++k) {
        float a = twopi * xn * freqs[k];
        f[1 + k]         = f2bf(sinf(a));
        f[1 + KFREQ + k] = f2bf(cosf(a));
    }
#pragma unroll
    for (int l = 0; l < LVLS; ++l) {
        int rm1 = res[l] - 1;
        float t = xn * (float)rm1;
        int i0 = (int)floorf(t);
        int i1 = i0 + 1; if (i1 > rm1) i1 = rm1;
        float w = t - (float)i0;
        uint32_t lt = (uint32_t)l * 19349663u;
        uint32_t h0 = (((uint32_t)i0 * 73856093u) ^ lt) & (HSIZE - 1);
        uint32_t h1 = (((uint32_t)i1 * 73856093u) ^ lt) & (HSIZE - 1);
        const float* e0 = ht + ((size_t)l * HSIZE + h0) * EDIM;
        const float* e1 = ht + ((size_t)l * HSIZE + h1) * EDIM;
#pragma unroll
        for (int e = 0; e < EDIM; ++e)
            f[1 + 2 * KFREQ + l * EDIM + e] = f2bf(e0[e] * (1.0f - w) + e1[e] * w);
    }
#pragma unroll
    for (int j = 0; j < LATD; ++j)
        f[1 + 2 * KFREQ + LVLS * EDIM + j] = f2bf(z[(size_t)b * LATD + j]);
    for (int j = INDIM; j < INPAD; ++j) f[j] = 0;
}

// ---------- WMMA GEMM, each wave computes a 16x64 strip (4 N-tiles, shared A frag) ----------
// out = act(A[M,K] @ Bt[N,K]^T + bias); grid must satisfy waves == (M/16)*(N/64).
__global__ void gemm_bias_act_kernel(const uint16_t* __restrict__ A, const uint16_t* __restrict__ Bt,
                                     const float* __restrict__ bias,
                                     float* outF, uint16_t* outB, int N, int K, int act) {
    int wave = (blockIdx.x * blockDim.x + threadIdx.x) >> 5;
    int lane = threadIdx.x & 31;
    int stripsN = N >> 6;                       // strips of 4 tiles
    int tm = wave / stripsN;
    int nt0 = (wave - tm * stripsN) << 2;       // first of 4 N-tiles
    v8f acc0 = {}, acc1 = {}, acc2 = {}, acc3 = {};
    for (int kk = 0; kk < K; kk += 32) {
        v16bf a = load_a_frag(A, tm << 4, K, kk);
        v16bf b0 = load_b_frag(Bt, (nt0 + 0) << 4, K, kk);
        v16bf b1 = load_b_frag(Bt, (nt0 + 1) << 4, K, kk);
        v16bf b2 = load_b_frag(Bt, (nt0 + 2) << 4, K, kk);
        v16bf b3 = load_b_frag(Bt, (nt0 + 3) << 4, K, kk);
        acc0 = __builtin_amdgcn_wmma_f32_16x16x32_bf16(false, a, false, b0, (short)0, acc0, false, false);
        acc1 = __builtin_amdgcn_wmma_f32_16x16x32_bf16(false, a, false, b1, (short)0, acc1, false, false);
        acc2 = __builtin_amdgcn_wmma_f32_16x16x32_bf16(false, a, false, b2, (short)0, acc2, false, false);
        acc3 = __builtin_amdgcn_wmma_f32_16x16x32_bf16(false, a, false, b3, (short)0, acc3, false, false);
    }
    int lh = lane >> 4;
    v8f accs[4] = {acc0, acc1, acc2, acc3};
#pragma unroll
    for (int j = 0; j < 4; ++j) {
        int n = ((nt0 + j) << 4) + (lane & 15);
        float bv = bias[n];
#pragma unroll
        for (int r = 0; r < 8; ++r) {
            int m = (tm << 4) + (lh << 3) + r;
            float v = accs[j][r] + bv;
            if (act) v = gelu_exact(v);
            size_t o = (size_t)m * N + n;
            if (outF) outF[o] = v;
            if (outB) outB[o] = f2bf(v);
        }
    }
}

// ---------- persistent fractal loop: 48 steps fully in LDS per 64-row tile ----------
__global__ void fractal_loop_kernel(const float* __restrict__ Hin, uint16_t* __restrict__ Hbf_out,
                                    const uint16_t* __restrict__ Wf1t, const float* __restrict__ bf1,
                                    const uint16_t* __restrict__ Wf2t, const float* __restrict__ bf2) {
    extern __shared__ char smem[];
    float*    Hf = (float*)smem;                                        // MB*HIDD fp32  (64 KB)
    uint16_t* Hb = (uint16_t*)(smem + MB * HIDD * 4);                   // MB*HIDD bf16  (32 KB)
    uint16_t* Gb = (uint16_t*)(smem + MB * HIDD * 4 + MB * HIDD * 2);   // MB*FFD  bf16  (64 KB)

    int tid  = threadIdx.x;
    int wave = tid >> 5;
    int lane = tid & 31;
    int lh   = lane >> 4;
    int row0 = blockIdx.x * MB;

    for (int i = tid; i < MB * HIDD; i += blockDim.x) {
        float v = Hin[(size_t)row0 * HIDD + i];
        Hf[i] = v;
        Hb[i] = f2bf(v);
    }
    __syncthreads();

    for (int t = 0; t < TSTEPS; ++t) {
        // phase 1: G = gelu(Hb @ Wf1 + bf1), [MB x FFD]
        // wave owns 4 consecutive N-tiles (shared A frag), looped over 4 M-tiles
#pragma unroll
        for (int mi = 0; mi < 4; ++mi) {
            int nt0 = wave << 2;
            v8f acc0 = {}, acc1 = {}, acc2 = {}, acc3 = {};
#pragma unroll
            for (int kk = 0; kk < HIDD; kk += 32) {
                v16bf a = load_a_frag(Hb, mi << 4, HIDD, kk);
                v16bf b0 = load_b_frag(Wf1t, (nt0 + 0) << 4, HIDD, kk);
                v16bf b1 = load_b_frag(Wf1t, (nt0 + 1) << 4, HIDD, kk);
                v16bf b2 = load_b_frag(Wf1t, (nt0 + 2) << 4, HIDD, kk);
                v16bf b3 = load_b_frag(Wf1t, (nt0 + 3) << 4, HIDD, kk);
                acc0 = __builtin_amdgcn_wmma_f32_16x16x32_bf16(false, a, false, b0, (short)0, acc0, false, false);
                acc1 = __builtin_amdgcn_wmma_f32_16x16x32_bf16(false, a, false, b1, (short)0, acc1, false, false);
                acc2 = __builtin_amdgcn_wmma_f32_16x16x32_bf16(false, a, false, b2, (short)0, acc2, false, false);
                acc3 = __builtin_amdgcn_wmma_f32_16x16x32_bf16(false, a, false, b3, (short)0, acc3, false, false);
            }
            v8f accs[4] = {acc0, acc1, acc2, acc3};
#pragma unroll
            for (int j = 0; j < 4; ++j) {
                int n = ((nt0 + j) << 4) + (lane & 15);
                float bv = bf1[n];
#pragma unroll
                for (int r = 0; r < 8; ++r) {
                    int m = (mi << 4) + (lh << 3) + r;
                    Gb[m * FFD + n] = f2bf(gelu_exact(accs[j][r] + bv));
                }
            }
        }
        __syncthreads();
        // phase 2: U = tanh(Gb @ Wf2 + bf2); Hf = 0.5*Hf + 0.5*U
#pragma unroll
        for (int mi = 0; mi < 4; ++mi) {
            int nt0 = wave << 1;
            v8f acc0 = {}, acc1 = {};
#pragma unroll
            for (int kk = 0; kk < FFD; kk += 32) {
                v16bf a = load_a_frag(Gb, mi << 4, FFD, kk);
                v16bf b0 = load_b_frag(Wf2t, (nt0 + 0) << 4, FFD, kk);
                v16bf b1 = load_b_frag(Wf2t, (nt0 + 1) << 4, FFD, kk);
                acc0 = __builtin_amdgcn_wmma_f32_16x16x32_bf16(false, a, false, b0, (short)0, acc0, false, false);
                acc1 = __builtin_amdgcn_wmma_f32_16x16x32_bf16(false, a, false, b1, (short)0, acc1, false, false);
            }
            v8f accs[2] = {acc0, acc1};
#pragma unroll
            for (int j = 0; j < 2; ++j) {
                int n = ((nt0 + j) << 4) + (lane & 15);
                float bv = bf2[n];
#pragma unroll
                for (int r = 0; r < 8; ++r) {
                    int m = (mi << 4) + (lh << 3) + r;
                    float u = tanhf(accs[j][r] + bv);
                    int idx = m * HIDD + n;
                    Hf[idx] = 0.5f * Hf[idx] + 0.5f * u;
                }
            }
        }
        __syncthreads();
        for (int i = tid; i < MB * HIDD; i += blockDim.x) Hb[i] = f2bf(Hf[i]);
        __syncthreads();
    }

    for (int i = tid; i < MB * HIDD; i += blockDim.x)
        Hbf_out[(size_t)row0 * HIDD + i] = Hb[i];
}

// ---------- host ----------
extern "C" void kernel_launch(void* const* d_in, const int* in_sizes, int n_in,
                              void* d_out, int out_size, void* d_ws, size_t ws_size,
                              hipStream_t stream) {
    const float* x    = (const float*)d_in[0];
    const float* z    = (const float*)d_in[1];
    const float* ht   = (const float*)d_in[2];
    const float* W1   = (const float*)d_in[3];
    const float* b1   = (const float*)d_in[4];
    const float* W2   = (const float*)d_in[5];
    const float* b2   = (const float*)d_in[6];
    const float* Wf1  = (const float*)d_in[7];
    const float* bf1  = (const float*)d_in[8];
    const float* Wf2  = (const float*)d_in[9];
    const float* bf2  = (const float*)d_in[10];
    const float* Wo   = (const float*)d_in[11];
    const float* bo   = (const float*)d_in[12];
    const int*   res  = (const int*)d_in[13];
    const float* frq  = (const float*)d_in[14];

    char* ws = (char*)d_ws;
    size_t off = 0;
    auto take = [&](size_t bytes) { size_t o = off; off = (off + bytes + 255) & ~(size_t)255; return o; };
    uint16_t* feats = (uint16_t*)(ws + take((size_t)BATCH * INPAD * 2));
    uint16_t* W1t   = (uint16_t*)(ws + take((size_t)HIDD * INPAD * 2));
    uint16_t* W2t   = (uint16_t*)(ws + take((size_t)HIDD * HIDD * 2));
    uint16_t* Wf1t  = (uint16_t*)(ws + take((size_t)FFD * HIDD * 2));
    uint16_t* Wf2t  = (uint16_t*)(ws + take((size_t)HIDD * FFD * 2));
    uint16_t* Wot   = (uint16_t*)(ws + take((size_t)NVOCAB * HIDD * 2));
    uint16_t* h1b   = (uint16_t*)(ws + take((size_t)BATCH * HIDD * 2));
    float*    Hf    = (float*)   (ws + take((size_t)BATCH * HIDD * 4));
    uint16_t* Hbf   = (uint16_t*)(ws + take((size_t)BATCH * HIDD * 2));
    (void)off; (void)ws_size; (void)in_sizes; (void)n_in; (void)out_size;

    const int TPB = 256;
    // weight prep (transpose + bf16 convert)
    wtrans_kernel<<<(HIDD * INPAD + TPB - 1) / TPB, TPB, 0, stream>>>(W1, W1t, INDIM, HIDD, INPAD);
    wtrans_kernel<<<(HIDD * HIDD + TPB - 1) / TPB, TPB, 0, stream>>>(W2, W2t, HIDD, HIDD, HIDD);
    wtrans_kernel<<<(FFD * HIDD + TPB - 1) / TPB, TPB, 0, stream>>>(Wf1, Wf1t, HIDD, FFD, HIDD);
    wtrans_kernel<<<(HIDD * FFD + TPB - 1) / TPB, TPB, 0, stream>>>(Wf2, Wf2t, FFD, HIDD, FFD);
    wtrans_kernel<<<(NVOCAB * HIDD + TPB - 1) / TPB, TPB, 0, stream>>>(Wo, Wot, HIDD, NVOCAB, HIDD);

    // feature build
    features_kernel<<<(BATCH + TPB - 1) / TPB, TPB, 0, stream>>>(x, z, ht, res, frq, feats);

    // h = gelu(feats @ W1 + b1)   -> bf16      (waves = 1024 * 4 = 4096 -> 512 blocks)
    {
        int waves = (BATCH / 16) * (HIDD / 64);
        gemm_bias_act_kernel<<<waves / 8, TPB, 0, stream>>>(feats, W1t, b1, nullptr, h1b, HIDD, INPAD, 1);
    }
    // h = gelu(h @ W2 + b2)       -> fp32 (loop state)
    {
        int waves = (BATCH / 16) * (HIDD / 64);
        gemm_bias_act_kernel<<<waves / 8, TPB, 0, stream>>>(h1b, W2t, b2, Hf, nullptr, HIDD, HIDD, 1);
    }
    // 48 fractal steps, persistent LDS tiles (160 KB LDS, 2 workgroups per WGP)
    {
        size_t shmem = (size_t)MB * HIDD * 4 + (size_t)MB * HIDD * 2 + (size_t)MB * FFD * 2;
        fractal_loop_kernel<<<BATCH / MB, TPB, shmem, stream>>>(Hf, Hbf, Wf1t, bf1, Wf2t, bf2);
    }
    // out = h @ Wo + bo           -> fp32
    {
        int waves = (BATCH / 16) * (NVOCAB / 64);
        gemm_bias_act_kernel<<<waves / 8, TPB, 0, stream>>>(Hbf, Wot, bo, (float*)d_out, nullptr, NVOCAB, HIDD, 0);
    }
}
